// SchNetLayer_31215822307957
// MI455X (gfx1250) — matrix operand order repaired and checked by reading, not compiled, not run on hardware
//
#include <hip/hip_runtime.h>

#define NN    10000
#define KK    32
#define FF    128
#define NRBFC 300
#define GAMMAC 10.0f

typedef __attribute__((ext_vector_type(16))) _Float16 v16h;
typedef __attribute__((ext_vector_type(8)))  float    v8f;

// ---- WMMA fragment math (layouts per CDNA5 ISA 7.12.2) ----------------------
// A (16x32 f16): lane holds row m=lane%16; half h -> k:
//   v=h/2,p=h&1; k = (v/4)*16 + hi*8 + (v%4)*2 + p      (hi = lane/16)
// inverse: given k: hi=(k>>3)&1, h = ((k>>4)&1)*8 + (k&7)
// B (32x16 f16): lane holds col n=lane%16; half h -> k = (h/8)*16 + hi*8 + h%8 (same as A)
// C/D (16x16 f32): vgpr r -> row r + hi*8, col = lane%16
__device__ __forceinline__ int a_kmap(int h, int hi) {
    int v = h >> 1, p = h & 1;
    return ((v >> 2) << 4) | (hi << 3) | ((v & 3) << 1) | p;
}

__device__ __forceinline__ v8f wmma16(v16h a, v16h b, v8f c) {
    return __builtin_amdgcn_wmma_f32_16x16x32_f16(false, a, false, b, (short)0, c,
                                                  false, false);
}

// shifted softplus: max(x,0)+log(1+exp(-|x|)) - ln2
__device__ __forceinline__ float ssp(float x) {
    return fmaxf(x, 0.0f) + __logf(1.0f + __expf(-fabsf(x))) - 0.69314718056f;
}

// A fragment from row-major f32 global: two contiguous 8-float runs per lane.
__device__ __forceinline__ v16h load_a_g32_vec(const float* __restrict__ A,
                                               int ld, int row0, int k0) {
    const int lane = threadIdx.x & 31;
    const int m = lane & 15, hi = lane >> 4;
    const float* r = A + (size_t)(row0 + m) * ld + k0 + hi * 8;
    float4 x0 = *(const float4*)(r + 0);
    float4 x1 = *(const float4*)(r + 4);
    float4 y0 = *(const float4*)(r + 16);
    float4 y1 = *(const float4*)(r + 20);
    v16h a;
    a[0] = (_Float16)x0.x;  a[1] = (_Float16)x0.y;
    a[2] = (_Float16)x0.z;  a[3] = (_Float16)x0.w;
    a[4] = (_Float16)x1.x;  a[5] = (_Float16)x1.y;
    a[6] = (_Float16)x1.z;  a[7] = (_Float16)x1.w;
    a[8] = (_Float16)y0.x;  a[9] = (_Float16)y0.y;
    a[10] = (_Float16)y0.z; a[11] = (_Float16)y0.w;
    a[12] = (_Float16)y1.x; a[13] = (_Float16)y1.y;
    a[14] = (_Float16)y1.z; a[15] = (_Float16)y1.w;
    return a;
}

// Packed B fragment fetch: 32B contiguous per lane -> 2x global_load_b128.
__device__ __forceinline__ v16h load_b_packed(const _Float16* __restrict__ pB,
                                              int kc, int nt) {
    const int lane = threadIdx.x & 31;
    return *(const v16h*)&pB[(size_t)((((kc << 3) | nt) << 5) + lane) << 4];
}

// ---- Kernel 0: pack W[Kdim x 128] f32 -> f16 B-fragment order ---------------
__global__ __launch_bounds__(256) void
pack_b_kernel(const float* __restrict__ W, int Kdim, int total,
              _Float16* __restrict__ dst) {
    int e = blockIdx.x * 256 + threadIdx.x;
    if (e >= total) return;
    int h = e & 15, ln = (e >> 4) & 31, nt = (e >> 9) & 7, kc = e >> 12;
    int k = kc * 32 + ((h >> 3) << 4) + ((ln >> 4) << 3) + (h & 7);
    int n = (nt << 4) | (ln & 15);
    float v = (k < Kdim) ? W[k * FF + n] : 0.0f;
    dst[e] = (_Float16)v;
}

// ---- Kernel 1: pre = atomic @ W_pre + b_pre --------------------------------
__global__ __launch_bounds__(256) void
schnet_pre_kernel(const float* __restrict__ atomic,
                  const _Float16* __restrict__ pWp,
                  const float* __restrict__ bp, float* __restrict__ pre) {
    const int lane = threadIdx.x & 31;
    const int w = threadIdx.x >> 5;
    const int n = lane & 15, hi = lane >> 4;
    const int row0 = blockIdx.x * 16;
    v8f acc = {};
#pragma unroll
    for (int kc = 0; kc < 4; ++kc) {
        v16h a = load_a_g32_vec(atomic, FF, row0, kc * 32);
        v16h b = load_b_packed(pWp, kc, w);
        acc = wmma16(a, b, acc);
    }
    const int col = (w << 4) | n;
    const float bb = bp[col];
#pragma unroll
    for (int r = 0; r < 8; ++r)
        pre[(row0 + r + hi * 8) * FF + col] = acc[r] + bb;
}

// ---- Kernel 2: fused rbf -> filter net -> masked K-reduction ---------------
// 2 nodes per workgroup (M = 64 edge rows), 128 threads = 4 waves.
// Wave w owns n-tiles {w, w+4}: each LDS A-fragment read feeds 2 WMMAs.
__global__ __launch_bounds__(128) void
schnet_cfconv_kernel(const float* __restrict__ xyz,
                     const float* __restrict__ edge_mask,
                     const int* __restrict__ src,
                     const _Float16* __restrict__ pBcf1,
                     const _Float16* __restrict__ pBcf2,
                     const float* __restrict__ centers,
                     const float* __restrict__ pre,
                     float* __restrict__ conv) {
    __shared__ float s_d[64];        // edge distances
    __shared__ float s_cent[320];    // padded centers: pad -> rbf underflows to 0
    __shared__ v16h  sAf[2][4][32];  // rbf A-fragments, double buffered (8KB)
    __shared__ v16h  sHf[4][4][32];  // hidden A-fragments [mt][kc][lane] (16KB)
    __shared__ float sP[64][132];    // pre[src]*mask (33KB)

    const int tid = threadIdx.x;
    const int lane = tid & 31;
    const int w = tid >> 5;          // 0..3
    const int n = lane & 15, hi = lane >> 4;
    const int i0 = blockIdx.x * 2;

    // padded center table: out-of-range centers make exp(-10 t^2) underflow to 0
    for (int j = tid; j < 320; j += 128)
        s_cent[j] = (j < NRBFC) ? centers[j] : 3.0e4f;

    // distances for 2 nodes x 32 neighbors
    if (tid < 64) {
        int node = i0 + (tid >> 5), k = tid & 31;
        int s = src[node * KK + k];
        float dx = xyz[3 * s + 0] - xyz[3 * node + 0];
        float dy = xyz[3 * s + 1] - xyz[3 * node + 1];
        float dz = xyz[3 * s + 2] - xyz[3 * node + 2];
        s_d[tid] = sqrtf(dx * dx + dy * dy + dz * dz);
    }
    // gather pre rows (L2-resident table), fold in mask; float4 granularity
    for (int e = tid; e < 64 * 32; e += 128) {
        int row = e >> 5, fq = (e & 31) << 2;
        int node = i0 + (row >> 5), k = row & 31;
        int s = src[node * KK + k];
        float m = edge_mask[node * KK + k];
        float4 pv = *(const float4*)&pre[(size_t)s * FF + fq];
        sP[row][fq + 0] = pv.x * m;
        sP[row][fq + 1] = pv.y * m;
        sP[row][fq + 2] = pv.z * m;
        sP[row][fq + 3] = pv.w * m;
    }

    // branchless cooperative rbf-chunk builder, writes in A-fragment order
    auto build = [&](int buf, int kc) {
        for (int e = tid; e < 4 * 32 * 16; e += 128) {
            int h = e & 15, ln = (e >> 4) & 31, mt = e >> 9;
            int row = (mt << 4) | (ln & 15);
            int j = kc * 32 + a_kmap(h, ln >> 4);
            float t = s_d[row] - s_cent[j];
            float u = t * (-GAMMAC);
            float v = __expf(t * u);       // v_exp, co-executes with XDL pipe
            ((_Float16*)&sAf[buf][mt][ln])[h] = (_Float16)v;
        }
    };

    __syncthreads();
    build(0, 0);
    __syncthreads();

    // GEMM1: H[64,128] = rbf[64,320] @ W_cf1 (zero-padded K), double buffered
    v8f hacc[4][2] = {};
    for (int kc = 0; kc < 10; ++kc) {
        int buf = kc & 1;
        if (kc < 9) {
            build(buf ^ 1, kc + 1);
            __builtin_prefetch(&pBcf1[(size_t)(((((kc + 1) << 3) | w) << 5) + lane) << 4], 0, 1);
        }
        v16h b0 = load_b_packed(pBcf1, kc, w);
        v16h b1 = load_b_packed(pBcf1, kc, w + 4);
#pragma unroll
        for (int mt = 0; mt < 4; ++mt) {
            v16h a = sAf[buf][mt][lane];
            hacc[mt][0] = wmma16(a, b0, hacc[mt][0]);
            hacc[mt][1] = wmma16(a, b1, hacc[mt][1]);
        }
        __syncthreads();
    }

    // ssp epilogue scatters straight into GEMM2's A-fragment layout
    {
        const int col0 = (w << 4) | n;          // col1 = col0 + 64
        const int kl = col0 & 31, kc2 = col0 >> 5;
        const int hi2 = (kl >> 3) & 1;
        const int h2 = ((kl >> 4) << 3) | (kl & 7);
#pragma unroll
        for (int mt = 0; mt < 4; ++mt) {
#pragma unroll
            for (int r = 0; r < 8; ++r) {
                int lane2 = ((r + hi * 8) & 15) | (hi2 << 4);
                ((_Float16*)&sHf[mt][kc2][lane2])[h2]     = (_Float16)ssp(hacc[mt][0][r]);
                ((_Float16*)&sHf[mt][kc2 + 2][lane2])[h2] = (_Float16)ssp(hacc[mt][1][r]);
            }
        }
    }
    __syncthreads();

    // GEMM2: filt[64,128] = ssp(H @ W_cf2)
    v8f facc[4][2] = {};
#pragma unroll
    for (int kc = 0; kc < 4; ++kc) {
        v16h b0 = load_b_packed(pBcf2, kc, w);
        v16h b1 = load_b_packed(pBcf2, kc, w + 4);
#pragma unroll
        for (int mt = 0; mt < 4; ++mt) {
            v16h a = sHf[mt][kc][lane];
            facc[mt][0] = wmma16(a, b0, facc[mt][0]);
            facc[mt][1] = wmma16(a, b1, facc[mt][1]);
        }
    }

    // conv[i, f] = sum_k ssp(filt) * pre[src]*mask ; lanes L/L+16 share a col
    const int col0 = (w << 4) | n;
    float p00 = 0.0f, p01 = 0.0f, p10 = 0.0f, p11 = 0.0f;
#pragma unroll
    for (int mt = 0; mt < 4; ++mt) {
        float aA = 0.0f, aB = 0.0f;
#pragma unroll
        for (int r = 0; r < 8; ++r) {
            int row = (mt << 4) + r + hi * 8;
            aA += ssp(facc[mt][0][r]) * sP[row][col0];
            aB += ssp(facc[mt][1][r]) * sP[row][col0 + 64];
        }
        if (mt < 2) { p00 += aA; p01 += aB; }
        else        { p10 += aA; p11 += aB; }
    }
    p00 += __shfl_xor(p00, 16, 32);
    p01 += __shfl_xor(p01, 16, 32);
    p10 += __shfl_xor(p10, 16, 32);
    p11 += __shfl_xor(p11, 16, 32);
    if (hi == 0) {
        conv[(size_t)(i0 + 0) * FF + col0]      = p00;
        conv[(size_t)(i0 + 0) * FF + col0 + 64] = p01;
        conv[(size_t)(i0 + 1) * FF + col0]      = p10;
        conv[(size_t)(i0 + 1) * FF + col0 + 64] = p11;
    }
}

// ---- Kernel 3: out = atomic + ssp(conv@W1 + b1)@W2 + b2 --------------------
__global__ __launch_bounds__(256) void
schnet_post_kernel(const float* __restrict__ conv,
                   const _Float16* __restrict__ pW1,
                   const float* __restrict__ b1,
                   const _Float16* __restrict__ pW2,
                   const float* __restrict__ b2,
                   const float* __restrict__ atomic, float* __restrict__ out) {
    __shared__ v16h sTf[4][32];  // intermediate in A-fragment order
    const int lane = threadIdx.x & 31;
    const int w = threadIdx.x >> 5;
    const int n = lane & 15, hi = lane >> 4;
    const int row0 = blockIdx.x * 16;
    const int col = (w << 4) | n;

    v8f acc = {};
#pragma unroll
    for (int kc = 0; kc < 4; ++kc) {
        v16h a = load_a_g32_vec(conv, FF, row0, kc * 32);
        v16h b = load_b_packed(pW1, kc, w);
        acc = wmma16(a, b, acc);
    }
    const float bb1 = b1[col];
    {
        const int kl = col & 31, kc2 = col >> 5;
        const int hi2 = (kl >> 3) & 1;
        const int h2 = ((kl >> 4) << 3) | (kl & 7);
#pragma unroll
        for (int r = 0; r < 8; ++r) {
            int lane2 = ((r + hi * 8) & 15) | (hi2 << 4);
            ((_Float16*)&sTf[kc2][lane2])[h2] = (_Float16)ssp(acc[r] + bb1);
        }
    }
    __syncthreads();

    v8f acc2 = {};
#pragma unroll
    for (int kc = 0; kc < 4; ++kc) {
        v16h b = load_b_packed(pW2, kc, w);
        acc2 = wmma16(sTf[kc][lane], b, acc2);
    }
    const float bb2 = b2[col];
#pragma unroll
    for (int r = 0; r < 8; ++r) {
        int row = row0 + r + hi * 8;
        out[(size_t)row * FF + col] = atomic[(size_t)row * FF + col] + acc2[r] + bb2;
    }
}

// ---- host launcher ---------------------------------------------------------
extern "C" void kernel_launch(void* const* d_in, const int* in_sizes, int n_in,
                              void* d_out, int out_size, void* d_ws, size_t ws_size,
                              hipStream_t stream) {
    const float* xyz       = (const float*)d_in[0];
    const float* atomic    = (const float*)d_in[1];
    const float* edge_mask = (const float*)d_in[2];
    const int*   src       = (const int*)d_in[3];
    const float* W_pre     = (const float*)d_in[4];
    const float* b_pre     = (const float*)d_in[5];
    const float* W_cf1     = (const float*)d_in[6];
    const float* W_cf2     = (const float*)d_in[7];
    const float* W_post1   = (const float*)d_in[8];
    const float* b_post1   = (const float*)d_in[9];
    const float* W_post2   = (const float*)d_in[10];
    const float* b_post2   = (const float*)d_in[11];
    const float* centers   = (const float*)d_in[12];
    float* out = (float*)d_out;

    // workspace layout
    float* pre  = (float*)d_ws;                   // [N, F] f32
    float* conv = pre + (size_t)NN * FF;          // [N, F] f32
    _Float16* pWpre = (_Float16*)(conv + (size_t)NN * FF);
    _Float16* pWcf1 = pWpre + 4 * 4096;           // packed fragments
    _Float16* pWcf2 = pWcf1 + 10 * 4096;
    _Float16* pWp1  = pWcf2 + 4 * 4096;
    _Float16* pWp2  = pWp1 + 4 * 4096;

    // per-call f32->f16 fragment packing of all weights (tiny)
    pack_b_kernel<<<4 * 16, 256, 0, stream>>>(W_pre, FF, 4 * 4096, pWpre);
    pack_b_kernel<<<10 * 16, 256, 0, stream>>>(W_cf1, NRBFC, 10 * 4096, pWcf1);
    pack_b_kernel<<<4 * 16, 256, 0, stream>>>(W_cf2, FF, 4 * 4096, pWcf2);
    pack_b_kernel<<<4 * 16, 256, 0, stream>>>(W_post1, FF, 4 * 4096, pWp1);
    pack_b_kernel<<<4 * 16, 256, 0, stream>>>(W_post2, FF, 4 * 4096, pWp2);

    schnet_pre_kernel<<<NN / 16, 256, 0, stream>>>(atomic, pWpre, b_pre, pre);
    schnet_cfconv_kernel<<<NN / 2, 128, 0, stream>>>(xyz, edge_mask, src, pWcf1,
                                                     pWcf2, centers, pre, conv);
    schnet_post_kernel<<<NN / 16, 256, 0, stream>>>(conv, pWp1, b_post1, pWp2,
                                                    b_post2, atomic, out);
}
